// Attention_39152921870938
// MI455X (gfx1250) — matrix operand order repaired and checked by reading, not compiled, not run on hardware
//
#include <hip/hip_runtime.h>
#include <hip/hip_bf16.h>
#include <math.h>

// ---------------------------------------------------------------------------
// Shapes (fixed by the reference)
// ---------------------------------------------------------------------------
#define D_MODEL 1024
#define D_QK    128
#define D_V     128
#define BATCH   4
#define SEQ     4096

typedef __attribute__((ext_vector_type(16))) __bf16 v16bf;
typedef __attribute__((ext_vector_type(8)))  __bf16 v8bf;
typedef __attribute__((ext_vector_type(8)))  float  v8f;

#define WMMA_BF16(a, b, c) \
    __builtin_amdgcn_wmma_f32_16x16x32_bf16(false, (a), false, (b), (short)0, (c), false, false)

__device__ __forceinline__ v8f vzero8() {
    v8f z;
#pragma unroll
    for (int i = 0; i < 8; ++i) z[i] = 0.0f;
    return z;
}

__device__ __forceinline__ v16bf make_a_frag(v8bf lo, v8bf hi) {
    v16bf a;
#pragma unroll
    for (int i = 0; i < 8; ++i) { a[i] = lo[i]; a[i + 8] = hi[i]; }
    return a;
}

// CDNA5 async copy: 16 bytes per lane, global -> LDS, tracked by ASYNCcnt.
__device__ __forceinline__ void async_copy16(unsigned ldsoff, const void* g) {
    asm volatile("global_load_async_to_lds_b128 %0, %1, off"
                 :: "v"(ldsoff), "v"(g) : "memory");
}
__device__ __forceinline__ void wait_async0() {
    asm volatile("s_wait_asynccnt 0x0" ::: "memory");
}
__device__ __forceinline__ unsigned lds_off(const void* p) {
    return (unsigned)(uintptr_t)p;   // LDS generic addr: low 32 bits = LDS offset
}

// ---------------------------------------------------------------------------
// Kernel 1: transpose + convert weights.  W[in=1024][out=128] fp32 ->
// Wt[m][out=128][in=1024] bf16 so WMMA B-fragments are K-contiguous.
// ---------------------------------------------------------------------------
__global__ __launch_bounds__(256) void wtrans_kernel(const float* __restrict__ Wq,
                                                     const float* __restrict__ Wk,
                                                     const float* __restrict__ Wv,
                                                     unsigned short* __restrict__ Wt_) {
    __bf16* Wt = (__bf16*)Wt_;
    int idx = blockIdx.x * blockDim.x + threadIdx.x;   // 3*128*1024 total
    int m = idx / (D_QK * D_MODEL);
    int r = idx % (D_QK * D_MODEL);
    int o = r / D_MODEL;
    int i = r % D_MODEL;
    const float* W = (m == 0) ? Wq : (m == 1) ? Wk : Wv;
    Wt[idx] = (__bf16)W[(size_t)i * D_QK + o];
}

// ---------------------------------------------------------------------------
// Kernel 2: projections.  Q/K row-major bf16, V transposed Vt[b][dv][seq].
// W k-slices async-staged into LDS (double-buffered); x prefetched across
// iterations; B fragments preloaded to decouple ds_load latency from WMMA.
// grid = (16384/128, 3), block = 256
// ---------------------------------------------------------------------------
#define WSH_LD 40   // padded row (32 used): 80B stride -> conflict-free reads

__global__ __launch_bounds__(256) void proj_kernel(const float* __restrict__ x,
                                                   const unsigned short* __restrict__ Wt_,
                                                   const float* __restrict__ bq,
                                                   const float* __restrict__ bk,
                                                   const float* __restrict__ bv,
                                                   unsigned short* __restrict__ Qb_,
                                                   unsigned short* __restrict__ Kb_,
                                                   unsigned short* __restrict__ Vt_) {
    __shared__ __align__(16) __bf16 Wsh[2][128][WSH_LD];

    const __bf16* Wt = (const __bf16*)Wt_;
    __bf16* Qb = (__bf16*)Qb_;
    __bf16* Kb = (__bf16*)Kb_;
    __bf16* Vt = (__bf16*)Vt_;

    const int m    = blockIdx.y;              // 0=Q, 1=K, 2=V
    const int tid  = threadIdx.x;
    const int wave = tid >> 5;
    const int lane = tid & 31;
    const int half = lane >> 4;
    const int l16  = lane & 15;
    const int r0   = blockIdx.x * 128 + wave * 16;

    const __bf16* wt   = Wt + (size_t)m * (D_QK * D_MODEL);
    const float*  bias = (m == 0) ? bq : (m == 1) ? bk : bv;

    auto issue_w = [&](int kk, int buf) {
#pragma unroll
        for (int c = 0; c < 2; ++c) {
            const int ch   = tid * 2 + c;         // 0..511
            const int col  = ch >> 2;             // 0..127
            const int part = ch & 3;              // 4 x 16B = 64B per col
            const __bf16* g = wt + (size_t)col * D_MODEL + kk + part * 8;
            async_copy16(lds_off(&Wsh[buf][col][part * 8]), g);
        }
    };

    v8f acc[8];
#pragma unroll
    for (int n = 0; n < 8; ++n) acc[n] = vzero8();

    const float* xrow = x + (size_t)(r0 + l16) * D_MODEL;

    // prologue: async W slice 0 + prefetch x chunk 0
    issue_w(0, 0);
    float4 f0 = *(const float4*)(xrow + half * 8);
    float4 f1 = *(const float4*)(xrow + half * 8 + 4);
    float4 f2 = *(const float4*)(xrow + half * 8 + 16);
    float4 f3 = *(const float4*)(xrow + half * 8 + 20);

    for (int kk = 0; kk < D_MODEL; kk += 32) {
        const int buf = (kk >> 5) & 1;
        wait_async0();
        __syncthreads();
        if (kk + 32 < D_MODEL) issue_w(kk + 32, buf ^ 1);

        // A fragment from prefetched x chunk
        v16bf a;
        a[0]  = (__bf16)f0.x; a[1]  = (__bf16)f0.y; a[2]  = (__bf16)f0.z; a[3]  = (__bf16)f0.w;
        a[4]  = (__bf16)f1.x; a[5]  = (__bf16)f1.y; a[6]  = (__bf16)f1.z; a[7]  = (__bf16)f1.w;
        a[8]  = (__bf16)f2.x; a[9]  = (__bf16)f2.y; a[10] = (__bf16)f2.z; a[11] = (__bf16)f2.w;
        a[12] = (__bf16)f3.x; a[13] = (__bf16)f3.y; a[14] = (__bf16)f3.z; a[15] = (__bf16)f3.w;

        // prefetch next x chunk (global loads overlap the WMMA burst)
        if (kk + 32 < D_MODEL) {
            const int kA = kk + 32 + half * 8;
            f0 = *(const float4*)(xrow + kA);
            f1 = *(const float4*)(xrow + kA + 4);
            f2 = *(const float4*)(xrow + kA + 16);
            f3 = *(const float4*)(xrow + kA + 20);
        }

        // preload all 8 B fragments, then burst the WMMAs
        v16bf bfr[8];
#pragma unroll
        for (int n = 0; n < 8; ++n)
            bfr[n] = make_a_frag(*(const v8bf*)&Wsh[buf][n * 16 + l16][half * 8],
                                 *(const v8bf*)&Wsh[buf][n * 16 + l16][half * 8 + 16]);
#pragma unroll
        for (int n = 0; n < 8; ++n)
            acc[n] = WMMA_BF16(a, bfr[n], acc[n]);
    }

    // Epilogue: bias, convert, scatter per C-fragment layout
#pragma unroll
    for (int n = 0; n < 8; ++n) {
        const int col  = n * 16 + l16;
        const float bc = bias[col];
#pragma unroll
        for (int v = 0; v < 8; ++v) {
            const int row = r0 + v + half * 8;
            const float val = acc[n][v] + bc;
            if (m == 2) {
                const int bb = row >> 12;
                const int ss = row & (SEQ - 1);
                Vt[((size_t)bb * D_V + col) * SEQ + ss] = (__bf16)val;
            } else {
                __bf16* dst = (m == 0) ? Qb : Kb;
                dst[(size_t)row * D_QK + col] = (__bf16)val;
            }
        }
    }
}

// ---------------------------------------------------------------------------
// Kernel 3: flash attention.  Async double-buffered K/V staging in LDS,
// fragment preloading, and row-sum-by-WMMA (P x ones) for the softmax
// denominator.  grid = (SEQ/128, BATCH), block = 256 (8 waves x 16 queries)
// ---------------------------------------------------------------------------
#define KSH_LD 136  // padded 128: 272B stride
#define VSH_LD 40   // padded 32:   80B stride
#define PSH_LD 40

__global__ __launch_bounds__(256) void attn_kernel(const unsigned short* __restrict__ Qb_,
                                                   const unsigned short* __restrict__ Kb_,
                                                   const unsigned short* __restrict__ Vt_,
                                                   float* __restrict__ out) {
    __shared__ __align__(16) __bf16 Ksh[2][32][KSH_LD];   // [key][dqk]
    __shared__ __align__(16) __bf16 Vsh[2][128][VSH_LD];  // [dv][key]
    __shared__ __align__(16) __bf16 Plds[8][16][PSH_LD];  // per-wave P tile

    const __bf16* Qb = (const __bf16*)Qb_;
    const __bf16* Kb = (const __bf16*)Kb_;
    const __bf16* Vt = (const __bf16*)Vt_;

    const int b    = blockIdx.y;
    const int tid  = threadIdx.x;
    const int wave = tid >> 5;
    const int lane = tid & 31;
    const int half = lane >> 4;
    const int l16  = lane & 15;
    const int q0   = blockIdx.x * 128 + wave * 16;

    const __bf16* Qbase = Qb + (size_t)b * SEQ * D_QK;
    const __bf16* Kbase = Kb + (size_t)b * SEQ * D_QK;
    const __bf16* Vbase = Vt + (size_t)b * D_V * SEQ;

    auto issue_kv = [&](int kb, int buf) {
        {   // K: 32 keys x 128B (dqk 0..63) = 256 x 16B chunks, 1 per thread
            const int key  = tid >> 3;
            const int part = tid & 7;
            const __bf16* g = Kbase + (size_t)(kb + key) * D_QK + part * 8;
            async_copy16(lds_off(&Ksh[buf][key][part * 8]), g);
        }
#pragma unroll
        for (int c = 0; c < 2; ++c) {   // V: 128 dv x 64B = 512 chunks, 2/thread
            const int ch   = tid * 2 + c;
            const int dv   = ch >> 2;
            const int part = ch & 3;
            const __bf16* g = Vbase + (size_t)dv * SEQ + kb + part * 8;
            async_copy16(lds_off(&Vsh[buf][dv][part * 8]), g);
        }
    };

    // Persistent Q fragments covering d_qk = 128
    v16bf qf[4];
    {
        const __bf16* qrow = Qbase + (size_t)(q0 + l16) * D_QK;
#pragma unroll
        for (int j = 0; j < 4; ++j) {
            const int kA = j * 32 + half * 8;
            v8bf lo = *(const v8bf*)(qrow + kA);
            v8bf hi = *(const v8bf*)(qrow + kA + 16);
            qf[j] = make_a_frag(lo, hi);
        }
    }

    // all-ones B fragment: P x ones = row sums, in C-fragment layout
    v16bf onesf;
#pragma unroll
    for (int i = 0; i < 16; ++i) onesf[i] = (__bf16)1.0f;

    v8f oacc[8];
#pragma unroll
    for (int n = 0; n < 8; ++n) oacc[n] = vzero8();
    float mrow[8], lrow[8];
#pragma unroll
    for (int v = 0; v < 8; ++v) { mrow[v] = -1.0e30f; lrow[v] = 0.0f; }

    const float cscale = 0.08838834764831845f * 1.4426950408889634f; // 1/sqrt(128)*log2(e)

    issue_kv(0, 0);
    for (int ib = 0; ib < SEQ / 32; ++ib) {
        const int buf = ib & 1;
        const int kb  = ib * 32;
        wait_async0();
        __syncthreads();
        if (ib + 1 < SEQ / 32) issue_kv(kb + 32, buf ^ 1);

        // ---- preload K fragments, burst the score WMMAs ----
        v16bf kfr[8];
#pragma unroll
        for (int j = 0; j < 4; ++j) {
            const int kB = j * 32 + half * 8;
            kfr[2 * j]     = make_a_frag(*(const v8bf*)&Ksh[buf][l16][kB],
                                         *(const v8bf*)&Ksh[buf][l16][kB + 16]);
            kfr[2 * j + 1] = make_a_frag(*(const v8bf*)&Ksh[buf][16 + l16][kB],
                                         *(const v8bf*)&Ksh[buf][16 + l16][kB + 16]);
        }
        v8f sf0 = vzero8();
        v8f sf1 = vzero8();
#pragma unroll
        for (int j = 0; j < 4; ++j) {
            sf0 = WMMA_BF16(qf[j], kfr[2 * j],     sf0);
            sf1 = WMMA_BF16(qf[j], kfr[2 * j + 1], sf1);
        }

        // ---- preload V fragments early: ds_loads retire under softmax VALU ----
        v16bf vfr[8];
#pragma unroll
        for (int n = 0; n < 8; ++n)
            vfr[n] = make_a_frag(*(const v8bf*)&Vsh[buf][n * 16 + l16][half * 8],
                                 *(const v8bf*)&Vsh[buf][n * 16 + l16][half * 8 + 16]);

        // ---- online softmax: row max via shuffles, probabilities to LDS ----
        float af[8];
#pragma unroll
        for (int v = 0; v < 8; ++v) {
            float s0 = sf0[v] * cscale;
            float s1 = sf1[v] * cscale;
            float tm = fmaxf(s0, s1);
#pragma unroll
            for (int mk = 1; mk < 16; mk <<= 1)
                tm = fmaxf(tm, __shfl_xor(tm, mk, 32));
            const float mnew  = fmaxf(mrow[v], tm);
            const float alpha = exp2f(mrow[v] - mnew);
            const float p0 = exp2f(s0 - mnew);
            const float p1 = exp2f(s1 - mnew);
            mrow[v] = mnew;
            af[v]   = alpha;
#pragma unroll
            for (int n = 0; n < 8; ++n) oacc[n][v] *= alpha;

            const int prow = v + half * 8;
            Plds[wave][prow][l16]      = (__bf16)p0;
            Plds[wave][prow][l16 + 16] = (__bf16)p1;
        }

        // wave-private C-layout -> A-layout transpose via LDS
        asm volatile("s_wait_dscnt 0x0" ::: "memory");
        v16bf pa;
        {
            const __bf16* pr = &Plds[wave][l16][0];
            v8bf lo = *(const v8bf*)(pr + half * 8);
            v8bf hi = *(const v8bf*)(pr + half * 8 + 16);
            pa = make_a_frag(lo, hi);
        }

        // ---- row sums by WMMA (P x ones), then PV burst ----
        v8f sacc = WMMA_BF16(pa, onesf, vzero8());
#pragma unroll
        for (int n = 0; n < 8; ++n)
            oacc[n] = WMMA_BF16(pa, vfr[n], oacc[n]);
#pragma unroll
        for (int v = 0; v < 8; ++v)
            lrow[v] = lrow[v] * af[v] + sacc[v];
    }

    // ---- epilogue ----
    float* obase = out + (size_t)b * SEQ * D_V;
#pragma unroll
    for (int n = 0; n < 8; ++n) {
        const int col = n * 16 + l16;
#pragma unroll
        for (int v = 0; v < 8; ++v) {
            const int row = q0 + v + half * 8;
            obase[(size_t)row * D_V + col] = oacc[n][v] / lrow[v];
        }
    }
}

// ---------------------------------------------------------------------------
// Host-side launcher
// ---------------------------------------------------------------------------
extern "C" void kernel_launch(void* const* d_in, const int* in_sizes, int n_in,
                              void* d_out, int out_size, void* d_ws, size_t ws_size,
                              hipStream_t stream) {
    const float* x  = (const float*)d_in[0];
    const float* Wq = (const float*)d_in[1];
    const float* bq = (const float*)d_in[2];
    const float* Wk = (const float*)d_in[3];
    const float* bk = (const float*)d_in[4];
    const float* Wv = (const float*)d_in[5];
    const float* bv = (const float*)d_in[6];
    float* out = (float*)d_out;

    char* ws = (char*)d_ws;
    const size_t WT_BYTES = (size_t)3 * D_QK * D_MODEL * 2;
    const size_t QK_BYTES = (size_t)BATCH * SEQ * D_QK * 2;
    unsigned short* Wt = (unsigned short*)(ws);
    unsigned short* Qb = (unsigned short*)(ws + WT_BYTES);
    unsigned short* Kb = (unsigned short*)(ws + WT_BYTES + QK_BYTES);
    unsigned short* Vt = (unsigned short*)(ws + WT_BYTES + 2 * QK_BYTES);

    {
        const int total = 3 * D_QK * D_MODEL;
        wtrans_kernel<<<total / 256, 256, 0, stream>>>(Wq, Wk, Wv, Wt);
    }
    {
        dim3 grid((BATCH * SEQ) / 128, 3);
        proj_kernel<<<grid, 256, 0, stream>>>(x, Wt, bq, bk, bv, Qb, Kb, Vt);
    }
    {
        dim3 grid(SEQ / 128, BATCH);
        attn_kernel<<<grid, 256, 0, stream>>>(Qb, Kb, Vt, out);
    }
}
